// SAGE3BN1MLP2_61426622267899
// MI455X (gfx1250) — matrix operand order repaired and checked by reading, not compiled, not run on hardware
//
#include <hip/hip_runtime.h>
#include <hip/hip_bf16.h>

// ---------------------------------------------------------------------------
// SAGE(3) + BN + mean-pool + MLP head for MI455X (gfx1250, wave32, WMMA).
// - Edge phase: fp32 storage + fp32 global atomics (h + agg are L2-resident:
//   51.2MB + 51.2MB < 192MB L2). This is the bandwidth-bound dominant phase.
// - GEMM phase: v_wmma_f32_16x16x32_f16, weights staged in LDS in
//   fragment-major layout (one 64KB copy per 8-wave workgroup, B-fragment
//   reads become single contiguous 32B ds_load accesses).
// ---------------------------------------------------------------------------

#define N_NODES   100000
#define N_EDGES   1600000
#define EMB       128
#define HID       128
#define N_CLASSES 10
#define N_GRAPHS  512
#define EPS       1e-5f

#define WAVES_PER_BLOCK 8
#define ROWS_PER_BLOCK  (WAVES_PER_BLOCK * 16)   // 128
#define W_HALVES        (HID * HID)              // 16384 halves = 32KB

typedef _Float16 v16h __attribute__((ext_vector_type(16)));
typedef float    v8f  __attribute__((ext_vector_type(8)));

// ---------------------------------------------------------------- utilities

__global__ void zero_kernel(float* __restrict__ p, int n) {
    int i = blockIdx.x * blockDim.x + threadIdx.x;
    if (i < n) p[i] = 0.0f;
}

// BatchNorm (eval): h = (x - mean) * rsqrt(var + eps) * gamma + beta
__global__ void bn_kernel(const float* __restrict__ x,
                          const float* __restrict__ gamma,
                          const float* __restrict__ beta,
                          const float* __restrict__ mean,
                          const float* __restrict__ var,
                          float* __restrict__ h, int n) {
    int i = blockIdx.x * blockDim.x + threadIdx.x;
    if (i >= n) return;
    int f = i & (EMB - 1);
    float inv = rsqrtf(var[f] + EPS);
    h[i] = (x[i] - mean[f]) * inv * gamma[f] + beta[f];
}

// deg[dst] += 1 per edge
__global__ void deg_kernel(const long long* __restrict__ dst,
                           float* __restrict__ deg, int nedges) {
    int e = blockIdx.x * blockDim.x + threadIdx.x;
    if (e < nedges) atomicAdd(&deg[(int)dst[e]], 1.0f);
}

__global__ void invdeg_kernel(const float* __restrict__ deg,
                              float* __restrict__ invdeg, int n) {
    int i = blockIdx.x * blockDim.x + threadIdx.x;
    if (i < n) invdeg[i] = 1.0f / fmaxf(deg[i], 1.0f);
}

// Edge scatter: one wave of 32 lanes per edge, 4 floats per lane (128 feats).
// agg[dst] += h[src]. fp32 atomics, resolved in L2 (agg is L2-resident).
__global__ void scatter_kernel(const float* __restrict__ h,
                               const long long* __restrict__ src,
                               const long long* __restrict__ dst,
                               float* __restrict__ agg, int nedges) {
    int t = blockIdx.x * blockDim.x + threadIdx.x;
    int e = t >> 5;
    int lane = t & 31;
    if (e >= nedges) return;
    size_t s = (size_t)src[e];
    size_t d = (size_t)dst[e];
    const float4* hp = (const float4*)(h + s * HID);
    float4 v = hp[lane];
    float* ap = agg + d * HID + lane * 4;
    atomicAdd(ap + 0, v.x);
    atomicAdd(ap + 1, v.y);
    atomicAdd(ap + 2, v.z);
    atomicAdd(ap + 3, v.w);
}

// W [K=128][N=128] f32 row-major -> f16 in WMMA-B *fragment-major* layout:
// fragment f = kb*8 + nt (kb = k/32, nt = n/16) is a contiguous 512-half
// (1KB) block; within it, lane L = ((k>>3)&1)*16 + (n&15) holds halves
// j = (k&7) | ((k>>1)&8) at f*512 + L*16 + j. A B-fragment read in the GEMM
// is then one aligned 32B LDS access per lane.
__global__ void wconv_kernel(const float* __restrict__ W,
                             _Float16* __restrict__ Wt) {
    int i = blockIdx.x * blockDim.x + threadIdx.x;
    if (i >= HID * HID) return;
    int k = i >> 7, n = i & 127;
    int kb = k >> 5, hi = (k >> 3) & 1, j = (k & 7) | ((k >> 1) & 8);
    int nt = n >> 4, nl = n & 15;
    int f = kb * 8 + nt;
    Wt[f * 512 + (hi * 16 + nl) * 16 + j] = (_Float16)W[k * HID + n];
}

// ------------------------------------------------------------- WMMA GEMM

// A-fragment (16x32 f16, MxK) per CDNA5 layout:
// lane L (r=L%16, hi=L/16): halves 0..7  = row r, K = 32*kb + 8*hi + 0..7
//                           halves 8..15 = row r, K = 32*kb + 16 + 8*hi + 0..7
__device__ __forceinline__ v16h load_a_frag(const float* __restrict__ rowp,
                                            int kb, int hi, float scale) {
    const float* p0 = rowp + kb * 32 + hi * 8;
    const float* p1 = p0 + 16;
    v16h a;
#pragma unroll
    for (int j = 0; j < 8; ++j) a[j] = (_Float16)(p0[j] * scale);
#pragma unroll
    for (int j = 0; j < 8; ++j) a[8 + j] = (_Float16)(p1[j] * scale);
    return a;
}

// out[128 rows x 128] = relu( (agg*invdeg) @ Wl + h @ Wr + bias )
// 8 waves per block; each wave owns a 16-row tile. Weights live in LDS
// (fragment-major), one cooperative 64KB copy per block.
__global__ void __launch_bounds__(256)
sage_gemm_kernel(const float* __restrict__ agg,
                 const float* __restrict__ invdeg,
                 const float* __restrict__ h,
                 const _Float16* __restrict__ WlT,
                 const _Float16* __restrict__ WrT,
                 const float* __restrict__ bias,
                 float* __restrict__ out) {
    __shared__ _Float16 shL[W_HALVES];
    __shared__ _Float16 shR[W_HALVES];

    // cooperative weight staging: 2 x 32KB = 2 x 2048 uint4
    {
        const uint4* gl = (const uint4*)WlT;
        const uint4* gr = (const uint4*)WrT;
        uint4* sl = (uint4*)shL;
        uint4* sr = (uint4*)shR;
#pragma unroll
        for (int it = 0; it < (W_HALVES / 8) / 256; ++it) {
            int i = it * 256 + threadIdx.x;
            sl[i] = gl[i];
            sr[i] = gr[i];
        }
    }
    __syncthreads();

    int lane = threadIdx.x & 31;
    int wave = threadIdx.x >> 5;
    int nodebase = blockIdx.x * ROWS_PER_BLOCK + wave * 16;
    int nl = lane & 15;              // column-within-tile / A-row select
    int hi = lane >> 4;

    int arow = nodebase + nl;
    if (arow > N_NODES - 1) arow = N_NODES - 1;   // clamp tail-block loads
    float scale = invdeg[arow];
    const float* aggrow = agg + (size_t)arow * HID;
    const float* hrow   = h   + (size_t)arow * HID;

    v8f acc[8];
#pragma unroll
    for (int nt = 0; nt < 8; ++nt) {
        float bv = bias[nt * 16 + nl];
#pragma unroll
        for (int r = 0; r < 8; ++r) acc[nt][r] = bv;
    }

#pragma unroll
    for (int kb = 0; kb < 4; ++kb) {
        v16h aAgg = load_a_frag(aggrow, kb, hi, scale);
        v16h aH   = load_a_frag(hrow,   kb, hi, 1.0f);
#pragma unroll
        for (int nt = 0; nt < 8; ++nt) {
            // fragment-major: one contiguous 32B LDS read per lane
            const v16h* pl = (const v16h*)(shL + (kb * 8 + nt) * 512 + lane * 16);
            const v16h* pr = (const v16h*)(shR + (kb * 8 + nt) * 512 + lane * 16);
            v16h bl = *pl;
            v16h br = *pr;
            acc[nt] = __builtin_amdgcn_wmma_f32_16x16x32_f16(
                false, aAgg, false, bl, (short)0, acc[nt], false, false);
            acc[nt] = __builtin_amdgcn_wmma_f32_16x16x32_f16(
                false, aH,   false, br, (short)0, acc[nt], false, false);
        }
    }

    // C/D layout: VGPR r -> row (r + 8*hi), col = ntile*16 + nl
#pragma unroll
    for (int nt = 0; nt < 8; ++nt) {
#pragma unroll
        for (int r = 0; r < 8; ++r) {
            int orow = nodebase + r + 8 * hi;
            if (orow < N_NODES) {
                float v = acc[nt][r];
                out[(size_t)orow * HID + nt * 16 + nl] = v > 0.0f ? v : 0.0f;
            }
        }
    }
}

// ------------------------------------------------------------- pooling/head

__global__ void counts_kernel(const long long* __restrict__ batch,
                              float* __restrict__ counts, int n) {
    int i = blockIdx.x * blockDim.x + threadIdx.x;
    if (i < n) atomicAdd(&counts[(int)batch[i]], 1.0f);
}

__global__ void poolsum_kernel(const float* __restrict__ h,
                               const long long* __restrict__ batch,
                               float* __restrict__ pooled, int nnodes) {
    int t = blockIdx.x * blockDim.x + threadIdx.x;
    int node = t >> 5;
    int lane = t & 31;
    if (node >= nnodes) return;
    size_t g = (size_t)batch[node];
    const float4* hp = (const float4*)(h + (size_t)node * HID);
    float4 v = hp[lane];
    float* pp = pooled + g * HID + lane * 4;
    atomicAdd(pp + 0, v.x);
    atomicAdd(pp + 1, v.y);
    atomicAdd(pp + 2, v.z);
    atomicAdd(pp + 3, v.w);
}

// g1 = (pooled / max(counts,1)) @ Wm1 + bm1        (512 x 128, no activation)
__global__ void mlp1_kernel(const float* __restrict__ pooled,
                            const float* __restrict__ counts,
                            const float* __restrict__ Wm1,
                            const float* __restrict__ bm1,
                            float* __restrict__ g1) {
    int i = blockIdx.x * blockDim.x + threadIdx.x;
    if (i >= N_GRAPHS * HID) return;
    int g = i >> 7, j = i & 127;
    float inv = 1.0f / fmaxf(counts[g], 1.0f);
    float acc = 0.0f;
    const float* prow = pooled + (size_t)g * HID;
#pragma unroll 4
    for (int k = 0; k < HID; ++k) acc += prow[k] * Wm1[k * HID + j];
    g1[i] = acc * inv + bm1[j];
}

// logits = g1 @ Wm2 + bm2; out = log_softmax(logits). One thread per graph.
__global__ void head_kernel(const float* __restrict__ g1,
                            const float* __restrict__ Wm2,
                            const float* __restrict__ bm2,
                            float* __restrict__ out) {
    int g = blockIdx.x * blockDim.x + threadIdx.x;
    if (g >= N_GRAPHS) return;
    const float* grow = g1 + (size_t)g * HID;
    float l[N_CLASSES];
#pragma unroll
    for (int c = 0; c < N_CLASSES; ++c) l[c] = bm2[c];
    for (int k = 0; k < HID; ++k) {
        float gv = grow[k];
#pragma unroll
        for (int c = 0; c < N_CLASSES; ++c) l[c] += gv * Wm2[k * N_CLASSES + c];
    }
    float m = l[0];
#pragma unroll
    for (int c = 1; c < N_CLASSES; ++c) m = fmaxf(m, l[c]);
    float s = 0.0f;
#pragma unroll
    for (int c = 0; c < N_CLASSES; ++c) s += expf(l[c] - m);
    float lse = m + logf(s);
#pragma unroll
    for (int c = 0; c < N_CLASSES; ++c) out[g * N_CLASSES + c] = l[c] - lse;
}

// ------------------------------------------------------------------ launch

extern "C" void kernel_launch(void* const* d_in, const int* in_sizes, int n_in,
                              void* d_out, int out_size, void* d_ws, size_t ws_size,
                              hipStream_t stream) {
    const float*     x     = (const float*)d_in[0];
    const long long* ei    = (const long long*)d_in[1];   // int64 [2, E]
    const long long* batch = (const long long*)d_in[2];   // int64 [N]
    const float* gamma = (const float*)d_in[3];
    const float* beta  = (const float*)d_in[4];
    const float* mean  = (const float*)d_in[5];
    const float* var   = (const float*)d_in[6];
    const float* Wl[3] = { (const float*)d_in[7],  (const float*)d_in[10], (const float*)d_in[13] };
    const float* Wr[3] = { (const float*)d_in[8],  (const float*)d_in[11], (const float*)d_in[14] };
    const float* bl[3] = { (const float*)d_in[9],  (const float*)d_in[12], (const float*)d_in[15] };
    const float* Wm1 = (const float*)d_in[16];
    const float* bm1 = (const float*)d_in[17];
    const float* Wm2 = (const float*)d_in[18];
    const float* bm2 = (const float*)d_in[19];
    float* out = (float*)d_out;

    const long long* src = ei;
    const long long* dst = ei + N_EDGES;

    // workspace layout (bytes)
    char* ws = (char*)d_ws;
    const size_t FEAT_BYTES = (size_t)N_NODES * HID * sizeof(float);   // 51.2 MB
    float*    hA     = (float*)(ws);
    float*    hB     = (float*)(ws + FEAT_BYTES);
    float*    agg    = (float*)(ws + 2 * FEAT_BYTES);
    float*    deg    = (float*)(ws + 3 * FEAT_BYTES);
    float*    invdeg = deg + N_NODES;
    _Float16* wl16   = (_Float16*)(invdeg + N_NODES);
    _Float16* wr16   = wl16 + W_HALVES;
    float*    pooled = (float*)(wr16 + W_HALVES);
    float*    counts = pooled + (size_t)N_GRAPHS * HID;
    float*    g1     = counts + N_GRAPHS;
    (void)ws_size; (void)n_in; (void)in_sizes; (void)out_size;

    const int TB = 256;
    const int nfeat = N_NODES * HID;                       // 12.8M

    // BatchNorm -> hA
    bn_kernel<<<(nfeat + TB - 1) / TB, TB, 0, stream>>>(x, gamma, beta, mean, var, hA, nfeat);

    // degrees (edges are layer-invariant)
    zero_kernel<<<(N_NODES + TB - 1) / TB, TB, 0, stream>>>(deg, N_NODES);
    deg_kernel<<<(N_EDGES + TB - 1) / TB, TB, 0, stream>>>(dst, deg, N_EDGES);
    invdeg_kernel<<<(N_NODES + TB - 1) / TB, TB, 0, stream>>>(deg, invdeg, N_NODES);

    const int scatter_blocks = (N_EDGES * 32 + TB - 1) / TB;            // 1 wave per edge
    const int gemm_blocks    = (N_NODES + ROWS_PER_BLOCK - 1) / ROWS_PER_BLOCK;  // 782

    float* hin  = hA;
    float* hout = hB;
    for (int layer = 0; layer < 3; ++layer) {
        zero_kernel<<<(nfeat + TB - 1) / TB, TB, 0, stream>>>(agg, nfeat);
        scatter_kernel<<<scatter_blocks, TB, 0, stream>>>(hin, src, dst, agg, N_EDGES);
        wconv_kernel<<<(HID * HID + TB - 1) / TB, TB, 0, stream>>>(Wl[layer], wl16);
        wconv_kernel<<<(HID * HID + TB - 1) / TB, TB, 0, stream>>>(Wr[layer], wr16);
        sage_gemm_kernel<<<gemm_blocks, 256, 0, stream>>>(agg, invdeg, hin, wl16, wr16, bl[layer], hout);
        float* t = hin; hin = hout; hout = t;
    }
    // after 3 swaps, final features are in `hin`

    // global mean pool
    zero_kernel<<<(N_GRAPHS * HID + TB - 1) / TB, TB, 0, stream>>>(pooled, N_GRAPHS * HID);
    zero_kernel<<<(N_GRAPHS + TB - 1) / TB, TB, 0, stream>>>(counts, N_GRAPHS);
    counts_kernel<<<(N_NODES + TB - 1) / TB, TB, 0, stream>>>(batch, counts, N_NODES);
    poolsum_kernel<<<(N_NODES * 32 + TB - 1) / TB, TB, 0, stream>>>(hin, batch, pooled, N_NODES);

    // MLP head + log_softmax
    mlp1_kernel<<<(N_GRAPHS * HID + TB - 1) / TB, TB, 0, stream>>>(pooled, counts, Wm1, bm1, g1);
    head_kernel<<<(N_GRAPHS + TB - 1) / TB, TB, 0, stream>>>(g1, Wm2, bm2, out);
}
